// VanillaRNNDiagonalMetric_63213328662550
// MI455X (gfx1250) — matrix operand order repaired and checked by reading, not compiled, not run on hardware
//
#include <hip/hip_runtime.h>
#include <hip/hip_bf16.h>

#define ALPHA 0.1f
#define HID 512
#define INP 128
#define OUTD 128
#define BAT 32
#define TSTEPS 4096

typedef __attribute__((ext_vector_type(16))) __bf16 v16bf;
typedef __attribute__((ext_vector_type(8)))  float  v8f;

union V16 { v16bf v; uint4 q[2]; __bf16 h[16]; };

// workspace layout (bytes)
#define WS_W   0u                      // W  fragments: 32nt*16ks*512 bf16 = 512KB
#define WS_HO  (512u * 1024u)          // Who fragments: 8nt*16ks*512 bf16 = 128KB
#define WS_U   (1024u * 1024u)         // u[t][b][h] bf16 = 4096*32*512*2 = 128MB
#define WS_EX  (136u * 1024u * 1024u)  // state exchange: 2 groups * 16*512 bf16 = 32KB

__device__ __forceinline__ float fast_tanh(float x) {
#if __has_builtin(__builtin_amdgcn_tanhf)
  return __builtin_amdgcn_tanhf(x);
#else
  return tanhf(x);
#endif
}

__device__ __forceinline__ void pack8(__bf16* h, float4 a, float4 b) {
  h[0]=(__bf16)a.x; h[1]=(__bf16)a.y; h[2]=(__bf16)a.z; h[3]=(__bf16)a.w;
  h[4]=(__bf16)b.x; h[5]=(__bf16)b.y; h[6]=(__bf16)b.z; h[7]=(__bf16)b.w;
}

// ---------------------------------------------------------------------------
// Phase 1a: pre-swizzle W = B * Phi[k]/Phi[n] and Who into WMMA B-fragment
// layout (32x16 bf16 per fragment; lanes 0-15 hold K 0..15, lanes 16-31 hold
// K 16..31, 16 bf16 contiguous per lane -> hot loop is pure ds_load_b128).
// ---------------------------------------------------------------------------
__global__ void prep_frags(const float* __restrict__ Phi, const float* __restrict__ Bm,
                           const float* __restrict__ Who, __bf16* __restrict__ wsW,
                           __bf16* __restrict__ wsHO) {
  int idx = blockIdx.x * 256 + threadIdx.x;
  if (idx < 32 * 16 * 512) {
    int frag = idx >> 9, e = idx & 511;
    int nt = frag >> 4, ks = frag & 15;
    int lane = e >> 4, q = e & 15;
    int k = ks * 32 + ((lane < 16) ? q : 16 + q);
    int n = nt * 16 + (lane & 15);
    // B-frag element (k,n) = W^T[k][n] = B[n][k] * Phi[k]/Phi[n]
    wsW[idx] = (__bf16)(Bm[n * HID + k] * Phi[k] / Phi[n]);
  } else {
    int i2 = idx - 32 * 16 * 512;
    if (i2 < 8 * 16 * 512) {
      int frag = i2 >> 9, e = i2 & 511;
      int nt = frag >> 4, ks = frag & 15;
      int lane = e >> 4, q = e & 15;
      int k = ks * 32 + ((lane < 16) ? q : 16 + q);
      int n = nt * 16 + (lane & 15);
      wsHO[i2] = (__bf16)Who[n * HID + k];   // Who^T[k][n] = Who[n][k]
    }
  }
}

// ---------------------------------------------------------------------------
// Phase 1b: u[t][b][h] = input[b][t][:] @ weight_ih^T + bias_hh, bf16 output.
// One (16-row, 16-col) WMMA tile per wave, K=128 in 4 x (16x16x32 bf16).
// ---------------------------------------------------------------------------
__global__ void __launch_bounds__(256) u_proj(const float* __restrict__ inp,
                                              const float* __restrict__ wih,
                                              const float* __restrict__ bhh,
                                              __bf16* __restrict__ wsU) {
  int tid = threadIdx.x;
  int w = tid >> 5, lane = tid & 31;
  int task = blockIdx.x * 8 + w;          // 8192 m-tiles * 32 n-tiles
  int nt = task & 31;
  int mt = task >> 5;
  int b = mt >> 8;
  int t0 = (mt & 255) << 4;
  int hiA = (lane & 16) ? 8 : 0;
  int hiB = (lane & 16) ? 8 : 0;
  const float* arow = inp + ((size_t)b * TSTEPS + t0 + (lane & 15)) * INP;
  int ncol = nt * 16 + (lane & 15);
  const float* brow = wih + (size_t)ncol * INP;
  v8f acc = {};
  for (int ks = 0; ks < 4; ++ks) {
    V16 a, bv;
    int kb = ks * 32 + hiA;
    pack8(a.h,     *(const float4*)(arow + kb),      *(const float4*)(arow + kb + 4));
    pack8(a.h + 8, *(const float4*)(arow + kb + 16), *(const float4*)(arow + kb + 20));
    int kb2 = ks * 32 + hiB;
    pack8(bv.h,     *(const float4*)(brow + kb2),     *(const float4*)(brow + kb2 + 4));
    pack8(bv.h + 8, *(const float4*)(brow + kb2 + 16), *(const float4*)(brow + kb2 + 20));
    acc = __builtin_amdgcn_wmma_f32_16x16x32_bf16(false, a.v, false, bv.v,
                                                  (short)0, acc, false, false);
  }
  float bb = bhh[ncol];
  int mhi = (lane & 16) ? 8 : 0;
#pragma unroll
  for (int r = 0; r < 8; ++r) {
    int m = r + mhi;
    size_t idx = ((size_t)(t0 + m) * BAT + b) * HID + ncol;  // [t][b][h]
    wsU[idx] = (__bf16)(acc[r] + bb);
  }
}

// ---------------------------------------------------------------------------
// Phase 2: serial scan. Grid (4,2): x = hidden slice (cluster of 4 WGs on one
// SE), y = batch group of 16 rows. 256 threads = 8 waves per WG.
//   wave w owns state columns [c*128 + w*16, +16): 16 WMMA/step (K=512)
//   + 4 WMMA of a K-quarter of hy = state@Who^T (LDS-reduced by waves 0/4).
// State master copy stays in f32 accumulator registers; bf16 copy in LDS feeds
// the A fragments. Cross-WG exchange: async LDS<->global at SE scope around a
// cluster split-barrier (s_barrier_signal/-wait -3). u(t+1) prefetched async.
// (Launch should request cluster dims {4,1,1}; without cluster launch the
//  -3 barriers degrade to S_NOP per ISA.)
// ---------------------------------------------------------------------------
__global__ void __launch_bounds__(256, 1)
rnn_scan(const __bf16* __restrict__ wsW, const __bf16* __restrict__ wsHO,
         const __bf16* __restrict__ wsU, __bf16* __restrict__ exch,
         const float* __restrict__ bias_oh, float* __restrict__ out) {
  extern __shared__ char smem[];
  __bf16* wlds   = (__bf16*)(smem);             // 131072 B : 8nt x 16ks W frags
  __bf16* holds  = (__bf16*)(smem + 131072);    //  32768 B : 2nt x 16ks Who frags
  __bf16* st     = (__bf16*)(smem + 163840);    //  16896 B : state 16 x (512 pad 528)
  __bf16* ubuf   = (__bf16*)(smem + 180736);    //   8192 B : 2 x 16x128 u tiles
  float*  partials = (float*)(smem + 188928);   //   8192 B : 8 waves x 16x16 f32

  const int c = blockIdx.x;        // hidden slice 0..3
  const int g = blockIdx.y;        // batch group 0..1
  const int tid = threadIdx.x;
  const int w = tid >> 5, lane = tid & 31;

  const uint32_t ldsW  = (uint32_t)(size_t)(void*)wlds;
  const uint32_t ldsHO = (uint32_t)(size_t)(void*)holds;
  const uint32_t stBase = (uint32_t)(size_t)(void*)st;
  const uint32_t ldsUb = (uint32_t)(size_t)(void*)ubuf;
  const uint64_t exBase = (uint64_t)(size_t)exch;

  // zero bf16 state (t=0 state is zero)
  for (int i = tid; i < (16 * 528 * 2) / 4; i += 256) ((uint32_t*)st)[i] = 0u;

  // prologue: async-load W slice (128KB), Who slice (32KB), u(t=0) into LDS
  {
    uint64_t sb = (uint64_t)(size_t)wsW + (uint64_t)c * 131072u;
    for (int i = 0; i < 32; ++i) {
      uint32_t off = (uint32_t)((w * 1024 + i * 32 + lane) * 16);
      uint32_t la = ldsW + off;
      asm volatile("global_load_async_to_lds_b128 %0, %1, %2"
                   :: "v"(la), "v"(off), "s"(sb) : "memory");
    }
  }
  {
    uint64_t sb = (uint64_t)(size_t)wsHO + (uint64_t)c * 32768u;
    for (int i = 0; i < 8; ++i) {
      uint32_t off = (uint32_t)((w * 256 + i * 32 + lane) * 16);
      uint32_t la = ldsHO + off;
      asm volatile("global_load_async_to_lds_b128 %0, %1, %2"
                   :: "v"(la), "v"(off), "s"(sb) : "memory");
    }
  }
  const int uitem = w * 32 + lane, urow = uitem >> 4, uch = uitem & 15;
  const uint32_t voffU = (uint32_t)(urow * 1024 + uch * 16);   // global: row stride 1024B
  const uint64_t ubase = (uint64_t)(size_t)wsU +
                         ((uint64_t)g * 16 * HID + (uint64_t)c * 128) * 2;
  {
    uint64_t sb = ubase;                                       // t = 0
    uint32_t la = ldsUb + (uint32_t)(urow * 256 + uch * 16);
    asm volatile("global_load_async_to_lds_b128 %0, %1, %2"
                 :: "v"(la), "v"(voffU), "s"(sb) : "memory");
  }
  asm volatile("s_wait_asynccnt 0x0" ::: "memory");
  __syncthreads();

  v8f s = {};                                 // persistent f32 state tile
  const int arow = lane & 15;                 // A-frag: M = lane%16
  const int asub = (lane & 16) ? 32 : 0;      // byte sub-offset for upper half-wave
  const int tile = w >> 2, qtr = w & 3;       // hy tile / K-quarter assignment
  const int mhi = (lane & 16) ? 8 : 0;
  const char* stp0 = (const char*)st + arow * 1056 + asub;
  const char* wp0 = (const char*)wlds + (size_t)w * 16 * 512 * 2 + lane * 32;
  const char* hp0 = (const char*)holds + (size_t)tile * 16 * 512 * 2 + lane * 32;

  for (int t = 0; t < TSTEPS; ++t) {
    v8f accW = {}, accHO = {};
#pragma unroll 4
    for (int ks = 0; ks < 16; ++ks) {
      V16 a, bw;
      a.q[0] = *(const uint4*)(stp0 + ks * 64);
      a.q[1] = *(const uint4*)(stp0 + ks * 64 + 16);
      bw.q[0] = *(const uint4*)(wp0 + ks * 1024);
      bw.q[1] = *(const uint4*)(wp0 + ks * 1024 + 16);
      accW = __builtin_amdgcn_wmma_f32_16x16x32_bf16(false, a.v, false, bw.v,
                                                     (short)0, accW, false, false);
      if ((ks >> 2) == qtr) {
        V16 bh;
        bh.q[0] = *(const uint4*)(hp0 + ks * 1024);
        bh.q[1] = *(const uint4*)(hp0 + ks * 1024 + 16);
        accHO = __builtin_amdgcn_wmma_f32_16x16x32_bf16(false, a.v, false, bh.v,
                                                        (short)0, accHO, false, false);
      }
    }
    // prefetch u(t+1) (overlaps epilogue + sync)
    if (t + 1 < TSTEPS) {
      uint64_t sb = ubase + (uint64_t)(t + 1) * 32768u;
      uint32_t la = ldsUb + (uint32_t)(((t + 1) & 1) * 4096 + urow * 256 + uch * 16);
      asm volatile("global_load_async_to_lds_b128 %0, %1, %2"
                   :: "v"(la), "v"(voffU), "s"(sb) : "memory");
    }
    __syncthreads();   // everyone done reading state_t from LDS

    // leaky-tanh update (state master copy stays in f32 registers)
    const __bf16* ub = ubuf + (t & 1) * 2048 + w * 16 + (lane & 15);
#pragma unroll
    for (int r = 0; r < 8; ++r) {
      float pre = accW[r] + (float)ub[(r + mhi) * 128];
      s[r] = s[r] + ALPHA * (fast_tanh(pre) - s[r]);
    }
    __bf16* sp = st + (size_t)(c * 128 + w * 16 + (lane & 15));
#pragma unroll
    for (int r = 0; r < 8; ++r) sp[(size_t)(r + mhi) * 528] = (__bf16)s[r];
    {
      float* pp = partials + w * 256 + lane * 8;
      *(float4*)pp       = make_float4(accHO[0], accHO[1], accHO[2], accHO[3]);
      *(float4*)(pp + 4) = make_float4(accHO[4], accHO[5], accHO[6], accHO[7]);
    }
    __syncthreads();

    // broadcast own state tile to cluster peers (SE scope)
    {
      int rr = lane & 15, ch = lane >> 4;
      uint32_t la = stBase + (uint32_t)(rr * 1056 + (c * 128 + w * 16 + ch * 8) * 2);
      uint32_t vo = (uint32_t)(rr * 1024 + ch * 16);
      uint64_t sb = exBase + (uint64_t)g * 16384u + (uint64_t)(c * 128 + w * 16) * 2;
      asm volatile("global_store_async_from_lds_b128 %0, %1, %2 scope:SCOPE_SE"
                   :: "v"(vo), "v"(la), "s"(sb) : "memory");
    }
    // hy_t = state_t @ Who^T + bias_oh (reduce the 4 K-quarters, waves 0 & 4)
    if ((w & 3) == 0) {
      int ocol = (2 * c + tile) * 16 + (lane & 15);
      float bo = bias_oh[ocol];
      const float* pb = partials + (w & 4) * 256 + lane * 8;
#pragma unroll
      for (int r = 0; r < 8; ++r) {
        float sum = bo + pb[r] + pb[256 + r] + pb[512 + r] + pb[768 + r];
        out[((size_t)(g * 16 + r + mhi) * TSTEPS + t) * OUTD + ocol] = sum;
      }
    }
    asm volatile("s_wait_asynccnt 0x0" ::: "memory");
    __syncthreads();
    if (w == 0) asm volatile("s_barrier_signal -3" ::: "memory");
    asm volatile("s_barrier_wait -3" ::: "memory");
    // gather the 3 peer slices into the local state copy (SE scope)
#pragma unroll
    for (int oi = 1; oi < 4; ++oi) {
      int o = (c + oi) & 3;
      uint32_t la = stBase + (uint32_t)(urow * 1056 + (o * 128 + uch * 8) * 2);
      uint64_t sb = exBase + (uint64_t)g * 16384u + (uint64_t)o * 256u;
      asm volatile("global_load_async_to_lds_b128 %0, %1, %2 scope:SCOPE_SE"
                   :: "v"(la), "v"(voffU), "s"(sb) : "memory");
    }
    asm volatile("s_wait_asynccnt 0x0" ::: "memory");
    __syncthreads();
  }
}

extern "C" void kernel_launch(void* const* d_in, const int* in_sizes, int n_in,
                              void* d_out, int out_size, void* d_ws, size_t ws_size,
                              hipStream_t stream) {
  (void)in_sizes; (void)n_in; (void)out_size; (void)ws_size;
  const float* inp = (const float*)d_in[0];
  const float* Phi = (const float*)d_in[1];
  const float* Bm  = (const float*)d_in[2];
  const float* wih = (const float*)d_in[3];
  const float* who = (const float*)d_in[4];
  const float* boh = (const float*)d_in[5];
  const float* bhh = (const float*)d_in[6];
  char* ws = (char*)d_ws;
  __bf16* wsW  = (__bf16*)(ws + WS_W);
  __bf16* wsHO = (__bf16*)(ws + WS_HO);
  __bf16* wsU  = (__bf16*)(ws + WS_U);
  __bf16* exch = (__bf16*)(ws + WS_EX);
  float* out = (float*)d_out;

  prep_frags<<<1280, 256, 0, stream>>>(Phi, Bm, who, wsW, wsHO);
  u_proj<<<32768, 256, 0, stream>>>(inp, wih, bhh, wsU);

  const int SMEM = 197120;  // 192.5 KB dynamic LDS (WGP supports 320 KB)
  hipFuncSetAttribute((const void*)rnn_scan,
                      hipFuncAttributeMaxDynamicSharedMemorySize, SMEM);
  rnn_scan<<<dim3(4, 2, 1), 256, SMEM, stream>>>(wsW, wsHO, wsU, exch, boh, out);
}
